// ThinAttentionBand_19215683682903
// MI455X (gfx1250) — compile-verified, matrix operand order checked by reading, not run on hardware
//
#include <hip/hip_runtime.h>
#include <hip/hip_bf16.h>
#include <math.h>
#include <stdint.h>

typedef __bf16 bf16;
typedef __attribute__((ext_vector_type(16))) __bf16 v16bf;
typedef __attribute__((ext_vector_type(8)))  float  v8f;
typedef __attribute__((ext_vector_type(4)))  unsigned int v4u;
typedef __attribute__((ext_vector_type(8)))  int v8i;
typedef __attribute__((ext_vector_type(4)))  int v4i;

#define DEV static __device__ __forceinline__

constexpr int kS   = 4096;
constexpr int kD   = 1024;
constexpr int kH   = 16;
constexpr int kHD  = 64;
constexpr int kL   = 3;
constexpr int kW   = 512;
constexpr int kNB  = kS / kW;   // 8
constexpr int kMLP = 4096;

#if defined(__gfx1250__) && __has_builtin(__builtin_amdgcn_tensor_load_to_lds)
#define HAVE_TDM 1
#endif

DEV int lane_id() { return threadIdx.x & 31; }

// ---------------------------------------------------------------------------
// WMMA bf16 fragment helpers (layouts per CDNA5 ISA 7.12.2, wave32).
// A (16x32 MxK) from a row-major [16][ld] tile: lane L holds row m=L%16 and
// two contiguous 8-element K runs (kg..kg+7, 16+kg..16+kg+7), kg=(L/16)*8.
// Contiguous per lane -> compiler merges into 2x ds_load_b128.
// The B operand uses the same pattern on a [N][K] (transposed) tile.
// ---------------------------------------------------------------------------
DEV v16bf load_frag_a(const bf16* __restrict__ src, int ld) {
  const int m  = lane_id() & 15;
  const int kg = (lane_id() >> 4) << 3;
  v16bf f;
#pragma unroll
  for (int v = 0; v < 8; ++v) {
    const int k = (v < 4) ? (kg + 2 * v) : (16 + kg + 2 * (v - 4));
    f[2 * v]     = src[(size_t)m * ld + k];
    f[2 * v + 1] = src[(size_t)m * ld + k + 1];
  }
  return f;
}

DEV v8f wmma_bf16(v16bf a, v16bf b, v8f c) {
  return __builtin_amdgcn_wmma_f32_16x16x32_bf16(
      /*neg_a=*/false, a, /*neg_b=*/false, b,
      /*c_mod=*/(short)0, c, /*reuse_a=*/false, /*reuse_b=*/false);
}

#if defined(HAVE_TDM)
// ---------------------------------------------------------------------------
// Tensor Data Mover: 2D tile (tile_rows x tile_k bf16) from a row-major
// [rows_total][row_len] tensor into LDS rows padded to (tile_k + 8) bf16
// (= 16 DWORDs data + 4 DWORDs pad -> pad_interval code 3, pad_amount code 3).
// D# layout per CDNA5 ISA.
// ---------------------------------------------------------------------------
DEV void tdm_load_tile(unsigned lds_off, const bf16* gsrc, int row_len,
                       int rows_total, int tile_k, int tile_rows) {
  const unsigned long long ga = (unsigned long long)(uintptr_t)gsrc;
  v4u g0;
  g0[0] = 1u;                                     // count=1, user desc
  g0[1] = lds_off;                                // lds_addr
  g0[2] = (unsigned)(ga & 0xFFFFFFFFu);           // global_addr[31:0]
  g0[3] = (unsigned)((ga >> 32) & 0x01FFFFFFu) | (2u << 30);  // addr hi | type=2
  v8i g1;
  g1[0] = (int)((1u << 16)      // data_size = 2 bytes
              | (1u << 20)      // pad_enable
              | (3u << 22)      // pad_interval: 16 DWORDs
              | (3u << 25));    // pad_amount:   4 DWORDs
  g1[1] = (int)(((unsigned)row_len & 0xFFFFu) << 16);          // tensor_dim0 lo
  g1[2] = (int)(((unsigned)row_len >> 16) |
                (((unsigned)rows_total & 0xFFFFu) << 16));     // dim0 hi|dim1 lo
  g1[3] = (int)(((unsigned)rows_total >> 16) |
                ((unsigned)tile_k << 16));                     // dim1 hi|tile_dim0
  g1[4] = tile_rows;                                           // tile_dim1
  g1[5] = row_len;                                             // dim0_stride lo
  g1[6] = 0;
  g1[7] = 0;
  v4i z4; z4[0] = z4[1] = z4[2] = z4[3] = 0;
#if __clang_major__ >= 23
  v8i z8;
#pragma unroll
  for (int i = 0; i < 8; ++i) z8[i] = 0;
  __builtin_amdgcn_tensor_load_to_lds(g0, g1, z4, z4, z8, 0);
#else
  __builtin_amdgcn_tensor_load_to_lds(g0, g1, z4, z4, 0);
#endif
}
#endif  // HAVE_TDM

// ---------------------------------------------------------------------------
// fp32 -> bf16 elementwise convert
// ---------------------------------------------------------------------------
__global__ void cvt_f32_bf16_kernel(const float* __restrict__ in,
                                    bf16* __restrict__ out, int n) {
  int i = blockIdx.x * 256 + threadIdx.x;
  if (i < n) out[i] = (bf16)in[i];
}

// ---------------------------------------------------------------------------
// fp32 [K][N] -> bf16 transposed [N][K], per-layer via blockIdx.z
// ---------------------------------------------------------------------------
__global__ __launch_bounds__(256)
void cvt_transpose_kernel(const float* __restrict__ in, bf16* __restrict__ out,
                          int K, int N) {
  __shared__ float tile[32][33];
  const size_t zoff = (size_t)blockIdx.z * K * N;
  const int bk = blockIdx.x * 32, bn = blockIdx.y * 32;
  const int tx = threadIdx.x & 31, ty = threadIdx.x >> 5;  // ty 0..7
#pragma unroll
  for (int i = ty; i < 32; i += 8)
    tile[i][tx] = in[zoff + (size_t)(bk + i) * N + bn + tx];
  __syncthreads();
#pragma unroll
  for (int i = ty; i < 32; i += 8)
    out[zoff + (size_t)(bn + i) * K + bk + tx] = (bf16)tile[tx][i];
}

// ---------------------------------------------------------------------------
// LayerNorm: one block per row of D=1024, 256 threads, 4 elems/thread
// ---------------------------------------------------------------------------
__global__ __launch_bounds__(256)
void layernorm_kernel(const float* __restrict__ x, const float* __restrict__ g,
                      const float* __restrict__ b, float* __restrict__ outf,
                      bf16* __restrict__ outb) {
  __shared__ float red[256];
  const int row = blockIdx.x, t = threadIdx.x;
  const float* xr = x + (size_t)row * kD;
  float v4[4];
  float s = 0.f;
#pragma unroll
  for (int i = 0; i < 4; ++i) { v4[i] = xr[t + 256 * i]; s += v4[i]; }
  red[t] = s; __syncthreads();
#pragma unroll
  for (int st = 128; st > 0; st >>= 1) {
    if (t < st) red[t] += red[t + st];
    __syncthreads();
  }
  const float mean = red[0] * (1.f / kD);
  __syncthreads();
  s = 0.f;
#pragma unroll
  for (int i = 0; i < 4; ++i) { float d = v4[i] - mean; s += d * d; }
  red[t] = s; __syncthreads();
#pragma unroll
  for (int st = 128; st > 0; st >>= 1) {
    if (t < st) red[t] += red[t + st];
    __syncthreads();
  }
  const float inv = rsqrtf(red[0] * (1.f / kD) + 1e-5f);
#pragma unroll
  for (int i = 0; i < 4; ++i) {
    const int c = t + 256 * i;
    const float y = (v4[i] - mean) * inv * g[c] + b[c];
    if (outf) outf[(size_t)row * kD + c] = y;
    if (outb) outb[(size_t)row * kD + c] = (bf16)y;
  }
}

// ---------------------------------------------------------------------------
// GEMM: out[M,N] = A_bf16[M,K] @ Bt_bf16[N,K]^T + bias (+res) (opt GELU)
// 128x128 block tile, 8 waves, each wave 32x64 = 2x4 WMMA frags, K-step 32.
// A and Bt tiles staged into LDS by the TDM (fallback: vectorized copies).
// ---------------------------------------------------------------------------
__global__ __launch_bounds__(256)
void gemm_bf16_kernel(const bf16* __restrict__ A, const bf16* __restrict__ Bt,
                      const float* __restrict__ bias,
                      const float* __restrict__ res,
                      float* __restrict__ outf, bf16* __restrict__ outb,
                      int M, int N, int K, int gelu) {
  __shared__ bf16 As[128][40];   // 80B rows: 64B data + 16B pad
  __shared__ bf16 Bs[128][40];

  const int m0 = blockIdx.x * 128, n0 = blockIdx.y * 128;
  const int t = threadIdx.x;
  const int wave = t >> 5;
  const int wm = wave >> 1;      // 0..3 -> +32 rows
  const int wn = wave & 1;       // 0..1 -> +64 cols

  v8f acc[2][4];
#pragma unroll
  for (int mi = 0; mi < 2; ++mi)
#pragma unroll
    for (int ni = 0; ni < 4; ++ni)
#pragma unroll
      for (int r = 0; r < 8; ++r) acc[mi][ni][r] = 0.f;

#if defined(HAVE_TDM)
  const unsigned as_off = (unsigned)(uintptr_t)&As[0][0];
  const unsigned bs_off = (unsigned)(uintptr_t)&Bs[0][0];
#else
  const int ra = t >> 1, ca = (t & 1) * 16;   // 128 rows x 32 staging
#endif

  for (int k0 = 0; k0 < K; k0 += 32) {
#if defined(HAVE_TDM)
    if (t < 32) {  // wave 0 issues both tile DMAs, drains TENSORcnt
      tdm_load_tile(as_off, A + (size_t)m0 * K + k0, K, M, 32, 128);
      tdm_load_tile(bs_off, Bt + (size_t)n0 * K + k0, K, N, 32, 128);
      __builtin_amdgcn_s_wait_tensorcnt(0);
    }
    __syncthreads();
#else
    {
      const bf16* g = A + (size_t)(m0 + ra) * K + k0 + ca;
      const uint4* gv = reinterpret_cast<const uint4*>(g);
      uint4* sv = reinterpret_cast<uint4*>(&As[ra][ca]);
      sv[0] = gv[0]; sv[1] = gv[1];
      if (k0 + 32 < K) __builtin_prefetch(g + 32, 0, 1);
    }
    {
      const bf16* g = Bt + (size_t)(n0 + ra) * K + k0 + ca;
      const uint4* gv = reinterpret_cast<const uint4*>(g);
      uint4* sv = reinterpret_cast<uint4*>(&Bs[ra][ca]);
      sv[0] = gv[0]; sv[1] = gv[1];
      if (k0 + 32 < K) __builtin_prefetch(g + 32, 0, 1);
    }
    __syncthreads();
#endif

    v16bf af[2], bfr[4];
#pragma unroll
    for (int mi = 0; mi < 2; ++mi)
      af[mi] = load_frag_a(&As[wm * 32 + mi * 16][0], 40);
#pragma unroll
    for (int ni = 0; ni < 4; ++ni)
      bfr[ni] = load_frag_a(&Bs[wn * 64 + ni * 16][0], 40);
#pragma unroll
    for (int mi = 0; mi < 2; ++mi)
#pragma unroll
      for (int ni = 0; ni < 4; ++ni)
        acc[mi][ni] = wmma_bf16(af[mi], bfr[ni], acc[mi][ni]);
    __syncthreads();
  }

  // epilogue (C layout: N = lane&15; M = r + 8*(lane>=16))
  const int ln = lane_id() & 15;
  const int mhi = (lane_id() >> 4) * 8;
#pragma unroll
  for (int mi = 0; mi < 2; ++mi)
#pragma unroll
    for (int ni = 0; ni < 4; ++ni)
#pragma unroll
      for (int r = 0; r < 8; ++r) {
        const int m = m0 + wm * 32 + mi * 16 + mhi + r;
        const int n = n0 + wn * 64 + ni * 16 + ln;
        float v = acc[mi][ni][r] + bias[n];
        if (gelu) v = 0.5f * v * (1.f + erff(v * 0.70710678f));
        if (res) v += res[(size_t)m * N + n];
        if (outf) outf[(size_t)m * N + n] = v;
        if (outb) outb[(size_t)m * N + n] = (bf16)v;
      }
}

// ---------------------------------------------------------------------------
// RoPE (half-split) on q,k + bf16 convert. q,k stay [S][H*HD]; v is written
// transposed as vbT[h][hd][s] so attention's P.V B-fragments are contiguous.
// ---------------------------------------------------------------------------
__global__ void rope_cvt_kernel(const float* __restrict__ qf,
                                const float* __restrict__ kf,
                                const float* __restrict__ vf,
                                const int* __restrict__ pos,
                                bf16* __restrict__ qb, bf16* __restrict__ kb,
                                bf16* __restrict__ vbT) {
  const int idx = blockIdx.x * 256 + threadIdx.x;
  if (idx >= kS * kH * 32) return;
  const int i = idx & 31;
  const int h = (idx >> 5) & (kH - 1);
  const int s = idx >> 9;
  const float p = (float)pos[s];
  const float freq = powf(10000.f, -(float)i / 32.f);
  const float a = p * freq;
  const float c = cosf(a), sn = sinf(a);
  const size_t base = (size_t)s * kD + (size_t)h * kHD;
  const float q1 = qf[base + i], q2 = qf[base + 32 + i];
  qb[base + i]      = (bf16)(q1 * c - q2 * sn);
  qb[base + 32 + i] = (bf16)(q2 * c + q1 * sn);
  const float k1 = kf[base + i], k2 = kf[base + 32 + i];
  kb[base + i]      = (bf16)(k1 * c - k2 * sn);
  kb[base + 32 + i] = (bf16)(k2 * c + k1 * sn);
  vbT[((size_t)h * kHD + i) * kS + s]      = (bf16)vf[base + i];
  vbT[((size_t)h * kHD + i + 32) * kS + s] = (bf16)vf[base + 32 + i];
}

// ---------------------------------------------------------------------------
// Flash sliding-window attention. Grid (NB, H, W/64). 128 threads = 4 waves,
// each wave owns 16 query rows. Keys for block nb: global [(nb-1)W, (nb+1)W).
// S = Q Kt via 2x WMMA (K=64 over HD); P.V via 1 A-frag x 4 B-frags (K=32).
// ---------------------------------------------------------------------------
__global__ __launch_bounds__(128)
void attn_kernel(const bf16* __restrict__ qb, const bf16* __restrict__ kb,
                 const bf16* __restrict__ vbT, bf16* __restrict__ ob) {
  __shared__ bf16 kt_s[32][72];   // [key][hd], 144B rows
  __shared__ bf16 vt_s[64][40];   // [hd][key], 80B rows (transposed V)
  __shared__ bf16 p_s[4][16][40];

  const int nb = blockIdx.x;
  const int h  = blockIdx.y;
  const int qc = blockIdx.z;
  const int t = threadIdx.x, wv = t >> 5;
  const int ln16 = lane_id() & 15;
  const int nhalf = lane_id() >> 4;
  const int qrow0 = qc * 64 + wv * 16;          // local to block nb

  const bf16* qbase = qb + (size_t)(nb * kW + qrow0) * kD + (size_t)h * kHD;
  const v16bf qf0 = load_frag_a(qbase, kD);
  const v16bf qf1 = load_frag_a(qbase + 32, kD);

  v8f o[4];
  float mrun[8], lrun[8];
#pragma unroll
  for (int f = 0; f < 4; ++f)
#pragma unroll
    for (int r = 0; r < 8; ++r) o[f][r] = 0.f;
#pragma unroll
  for (int r = 0; r < 8; ++r) { mrun[r] = -INFINITY; lrun[r] = 0.f; }

  const int rk = t >> 2, ck = (t & 3) * 16;     // K staging: 32 keys x 64
  const int rv = t >> 1, cv = (t & 1) * 16;     // V staging: 64 hd x 32 keys

  for (int kt = 0; kt < 32; ++kt) {
    const int j0 = (nb - 1) * kW + kt * 32;     // first global key of tile
    if (j0 + 32 <= 0) continue;                  // fully before t=0 (uniform)

    {  // stage K tile [key][hd] (all j in range after the skip above)
      const bf16* gk = kb + (size_t)(j0 + rk) * kD + (size_t)h * kHD + ck;
      reinterpret_cast<uint4*>(&kt_s[rk][ck])[0] =
          reinterpret_cast<const uint4*>(gk)[0];
      reinterpret_cast<uint4*>(&kt_s[rk][ck])[1] =
          reinterpret_cast<const uint4*>(gk)[1];
    }
    {  // stage V tile transposed [hd][key] from vbT[h][hd][s]
      const bf16* gv = vbT + ((size_t)h * kHD + rv) * kS + j0 + cv;
      reinterpret_cast<uint4*>(&vt_s[rv][cv])[0] =
          reinterpret_cast<const uint4*>(gv)[0];
      reinterpret_cast<uint4*>(&vt_s[rv][cv])[1] =
          reinterpret_cast<const uint4*>(gv)[1];
    }
    __syncthreads();

    // scores: two 16x16 key sub-tiles, each accumulated over HD=64
    v8f s0, s1;
#pragma unroll
    for (int r = 0; r < 8; ++r) { s0[r] = 0.f; s1[r] = 0.f; }
    s0 = wmma_bf16(qf0, load_frag_a(&kt_s[0][0], 72), s0);
    s0 = wmma_bf16(qf1, load_frag_a(&kt_s[0][32], 72), s0);
    s1 = wmma_bf16(qf0, load_frag_a(&kt_s[16][0], 72), s1);
    s1 = wmma_bf16(qf1, load_frag_a(&kt_s[16][32], 72), s1);

    float corr[8];
#pragma unroll
    for (int r = 0; r < 8; ++r) {
      const int mloc = nhalf * 8 + r;
      const int iglob = nb * kW + qrow0 + mloc;
      const int jA = j0 + ln16, jB = jA + 16;
      float v0 = s0[r] * 0.125f;   // HD^-0.5
      float v1 = s1[r] * 0.125f;
      const int dA = iglob - jA, dB = iglob - jB;
      v0 = (dA >= 0 && dA < kW) ? v0 : -INFINITY;
      v1 = (dB >= 0 && dB < kW) ? v1 : -INFINITY;
      float tm = fmaxf(v0, v1);
#pragma unroll
      for (int off = 8; off >= 1; off >>= 1) tm = fmaxf(tm, __shfl_xor(tm, off));
      const float mn = fmaxf(mrun[r], tm);
      const float msafe = (mn == -INFINITY) ? 0.f : mn;
      const float c = __expf(mrun[r] - msafe);
      const float e0 = __expf(v0 - msafe);
      const float e1 = __expf(v1 - msafe);
      float rsum = e0 + e1;
#pragma unroll
      for (int off = 8; off >= 1; off >>= 1) rsum += __shfl_xor(rsum, off);
      lrun[r] = lrun[r] * c + rsum;
      mrun[r] = mn;
      corr[r] = c;
      p_s[wv][mloc][ln16]      = (bf16)e0;
      p_s[wv][mloc][16 + ln16] = (bf16)e1;
    }
#pragma unroll
    for (int f = 0; f < 4; ++f)
#pragma unroll
      for (int r = 0; r < 8; ++r) o[f][r] *= corr[r];

    // O += P(16x32) @ V(32x64); V B-frags from transposed [hd][key] tile
    const v16bf pf = load_frag_a(&p_s[wv][0][0], 40);
#pragma unroll
    for (int f = 0; f < 4; ++f)
      o[f] = wmma_bf16(pf, load_frag_a(&vt_s[f * 16][0], 40), o[f]);
    __syncthreads();
  }

#pragma unroll
  for (int f = 0; f < 4; ++f)
#pragma unroll
    for (int r = 0; r < 8; ++r) {
      const int mloc = nhalf * 8 + r;
      const float denom = (lrun[r] > 0.f) ? lrun[r] : 1.f;
      const size_t idx = (size_t)(nb * kW + qrow0 + mloc) * kD +
                         (size_t)h * kHD + f * 16 + ln16;
      ob[idx] = (bf16)(o[f][r] / denom);
    }
}

// ---------------------------------------------------------------------------
// Host-side orchestration
// ---------------------------------------------------------------------------
extern "C" void kernel_launch(void* const* d_in, const int* in_sizes, int n_in,
                              void* d_out, int out_size, void* d_ws,
                              size_t ws_size, hipStream_t stream) {
  (void)in_sizes; (void)n_in; (void)out_size; (void)ws_size;
  const float* hs    = (const float*)d_in[0];
  const int*   pos   = (const int*)d_in[1];
  const float* ln1_g = (const float*)d_in[2];
  const float* ln1_b = (const float*)d_in[3];
  const float* wq    = (const float*)d_in[4];
  const float* bq    = (const float*)d_in[5];
  const float* wk    = (const float*)d_in[6];
  const float* bk    = (const float*)d_in[7];
  const float* wv    = (const float*)d_in[8];
  const float* bv    = (const float*)d_in[9];
  const float* wo    = (const float*)d_in[10];
  const float* bo    = (const float*)d_in[11];
  const float* ln2_g = (const float*)d_in[12];
  const float* ln2_b = (const float*)d_in[13];
  const float* w1    = (const float*)d_in[14];
  const float* b1    = (const float*)d_in[15];
  const float* w2    = (const float*)d_in[16];
  const float* b2    = (const float*)d_in[17];
  const float* fln_g = (const float*)d_in[18];
  const float* fln_b = (const float*)d_in[19];

  char* ws = (char*)d_ws;
  size_t off = 0;
  auto carve = [&](size_t bytes) -> void* {
    void* p = ws + off;
    off += (bytes + 255) & ~(size_t)255;
    return p;
  };
  const size_t SD = (size_t)kS * kD;
  float* x  = (float*)carve(SD * 4);
  float* qf = (float*)carve(SD * 4);
  float* kf = (float*)carve(SD * 4);
  float* vf = (float*)carve(SD * 4);
  bf16* hb   = (bf16*)carve(SD * 2);
  bf16* qb   = (bf16*)carve(SD * 2);
  bf16* kb   = (bf16*)carve(SD * 2);
  bf16* vbT  = (bf16*)carve(SD * 2);
  bf16* obuf = (bf16*)carve(SD * 2);
  bf16* mb   = (bf16*)carve((size_t)kS * kMLP * 2);
  const int nW  = kL * kD * kD;
  const int nW1 = kL * kD * kMLP;
  bf16* wqb = (bf16*)carve((size_t)nW * 2);   // all weights stored transposed
  bf16* wkb = (bf16*)carve((size_t)nW * 2);
  bf16* wvb = (bf16*)carve((size_t)nW * 2);
  bf16* wob = (bf16*)carve((size_t)nW * 2);
  bf16* w1b = (bf16*)carve((size_t)nW1 * 2);
  bf16* w2b = (bf16*)carve((size_t)nW1 * 2);

  // weights -> bf16, transposed to [N][K] per layer
  const dim3 tDD(kD / 32, kD / 32, kL);
  cvt_transpose_kernel<<<tDD, 256, 0, stream>>>(wq, wqb, kD, kD);
  cvt_transpose_kernel<<<tDD, 256, 0, stream>>>(wk, wkb, kD, kD);
  cvt_transpose_kernel<<<tDD, 256, 0, stream>>>(wv, wvb, kD, kD);
  cvt_transpose_kernel<<<tDD, 256, 0, stream>>>(wo, wob, kD, kD);
  const dim3 tDM(kD / 32, kMLP / 32, kL);
  cvt_transpose_kernel<<<tDM, 256, 0, stream>>>(w1, w1b, kD, kMLP);
  const dim3 tMD(kMLP / 32, kD / 32, kL);
  cvt_transpose_kernel<<<tMD, 256, 0, stream>>>(w2, w2b, kMLP, kD);

  hipMemcpyAsync(x, hs, SD * 4, hipMemcpyDeviceToDevice, stream);

  const dim3 gN1024(kS / 128, kD / 128);    // 32 x 8
  const dim3 gN4096(kS / 128, kMLP / 128);  // 32 x 32
  const int ropeBlocks = (kS * kH * 32 + 255) / 256;

  for (int l = 0; l < kL; ++l) {
    const size_t wOff  = (size_t)l * kD * kD;
    const size_t w1Off = (size_t)l * kD * kMLP;
    const size_t bD    = (size_t)l * kD;
    const size_t bM    = (size_t)l * kMLP;

    layernorm_kernel<<<kS, 256, 0, stream>>>(x, ln1_g + bD, ln1_b + bD,
                                             nullptr, hb);
    gemm_bf16_kernel<<<gN1024, 256, 0, stream>>>(hb, wqb + wOff, bq + bD,
        nullptr, qf, nullptr, kS, kD, kD, 0);
    gemm_bf16_kernel<<<gN1024, 256, 0, stream>>>(hb, wkb + wOff, bk + bD,
        nullptr, kf, nullptr, kS, kD, kD, 0);
    gemm_bf16_kernel<<<gN1024, 256, 0, stream>>>(hb, wvb + wOff, bv + bD,
        nullptr, vf, nullptr, kS, kD, kD, 0);
    rope_cvt_kernel<<<ropeBlocks, 256, 0, stream>>>(qf, kf, vf, pos,
                                                    qb, kb, vbT);
    attn_kernel<<<dim3(kNB, kH, kW / 64), 128, 0, stream>>>(qb, kb, vbT, obuf);
    gemm_bf16_kernel<<<gN1024, 256, 0, stream>>>(obuf, wob + wOff, bo + bD,
        x, x, nullptr, kS, kD, kD, 0);                     // x += o@wo + bo
    layernorm_kernel<<<kS, 256, 0, stream>>>(x, ln2_g + bD, ln2_b + bD,
                                             nullptr, hb);
    gemm_bf16_kernel<<<gN4096, 256, 0, stream>>>(hb, w1b + w1Off, b1 + bM,
        nullptr, nullptr, mb, kS, kMLP, kD, 1);            // gelu -> bf16
    gemm_bf16_kernel<<<gN1024, 256, 0, stream>>>(mb, w2b + w1Off, b2 + bD,
        x, x, nullptr, kS, kD, kMLP, 0);                   // x += m@w2 + b2
  }
  layernorm_kernel<<<kS, 256, 0, stream>>>(x, fln_g, fln_b,
                                           (float*)d_out, nullptr);
}
